// CBOW_17274358464869
// MI455X (gfx1250) — matrix operand order repaired and checked by reading, not compiled, not run on hardware
//
#include <hip/hip_runtime.h>

typedef __attribute__((ext_vector_type(16))) _Float16 v16h;
typedef __attribute__((ext_vector_type(8)))  float    v8f;
typedef __attribute__((ext_vector_type(4)))  float    f4;

namespace {
constexpr int kD = 128;       // embedding dim
constexpr int kC = 10;        // ctx slots
constexpr int kN = 5;         // negatives
constexpr int kB = 131072;    // batch
constexpr int kTiles = kB / 16;          // 8192 wave-tiles, 16 elements each
constexpr int kWavesPerBlock = 8;
constexpr int kBlock = kWavesPerBlock * 32;   // wave32
}

// -log_sigmoid(clip(x, -10, 10)) = log1p(exp(-clip(x)))
__device__ __forceinline__ float nls_clip(float x) {
    x = fminf(fmaxf(x, -10.0f), 10.0f);
    return log1pf(__expf(-x));
}

// runtime-indexed extract from a v8f accumulator (lowers to v_cndmask chain)
__device__ __forceinline__ float pick8(v8f c, int r) {
    float v = c[0];
    if (r == 1) v = c[1];
    if (r == 2) v = c[2];
    if (r == 3) v = c[3];
    if (r == 4) v = c[4];
    if (r == 5) v = c[5];
    if (r == 6) v = c[6];
    if (r == 7) v = c[7];
    return v;
}

__global__ void __launch_bounds__(kBlock)
cbow_loss_kernel(const float* __restrict__ emb0,
                 const float* __restrict__ emb1,
                 const int*   __restrict__ word_idx,
                 const int*   __restrict__ ctx_inds,
                 const float* __restrict__ ctx_lens,
                 const int*   __restrict__ neg_inds,
                 float*       __restrict__ partials) {
    const int lane = threadIdx.x & 31;
    const int wid  = threadIdx.x >> 5;
    const int tile = blockIdx.x * kWavesPerBlock + wid;
    const int m    = lane & 15;     // element within the 16-wide tile
    const int hi   = lane >> 4;     // lanes m and m+16 split the K range
    const int e    = tile * 16 + m; // global batch element

    // Context sum held directly in WMMA A-operand order:
    // cs[t][g][j] = sum_c emb0[ctx[e][c]][ t*32 + g*16 + hi*8 + j ]
    float cs[4][2][8];
    #pragma unroll
    for (int t = 0; t < 4; ++t)
        #pragma unroll
        for (int g = 0; g < 2; ++g)
            #pragma unroll
            for (int j = 0; j < 8; ++j) cs[t][g][j] = 0.0f;

    #pragma unroll 2
    for (int c = 0; c < kC; ++c) {
        const int idx = ctx_inds[(size_t)e * kC + c];      // PAD row in emb0 is zeroed
        const float* row = emb0 + (size_t)idx * kD;
        #pragma unroll
        for (int t = 0; t < 4; ++t) {
            #pragma unroll
            for (int g = 0; g < 2; ++g) {
                const f4* p = (const f4*)(row + t * 32 + g * 16 + hi * 8);
                f4 v0 = p[0];
                f4 v1 = p[1];
                #pragma unroll
                for (int j = 0; j < 4; ++j) {
                    cs[t][g][j]     += v0[j];
                    cs[t][g][4 + j] += v1[j];
                }
            }
        }
    }

    // CBOW mean -> f16 A operand (16x32 per K-chunk, 4 chunks cover D=128)
    const float invLen = 1.0f / ctx_lens[e];
    v16h a[4];
    #pragma unroll
    for (int t = 0; t < 4; ++t)
        #pragma unroll
        for (int j = 0; j < 8; ++j) {
            a[t][j]     = (_Float16)(cs[t][0][j] * invLen);  // K = 32t + hi*8 + j
            a[t][j + 8] = (_Float16)(cs[t][1][j] * invLen);  // K = 32t + 16 + hi*8 + j
        }

    // Diagonal ownership: element m<8 -> lane m, vgpr m; m>=8 -> lane m+16, vgpr m-8
    const bool own = (lane < 8) || (lane >= 24);
    const int  r   = (lane < 8) ? lane : (lane - 24);

    float loss = 0.0f;

    // Target 0 = positive word, targets 1..5 = negatives. One WMMA chain each.
    #pragma unroll
    for (int tgt = 0; tgt < 6; ++tgt) {
        const int idx = (tgt == 0) ? word_idx[e]
                                   : neg_inds[(size_t)e * kN + (tgt - 1)];
        const float* row = emb1 + (size_t)idx * kD;

        v8f acc = {};
        #pragma unroll
        for (int t = 0; t < 4; ++t) {
            // B operand 32x16: lane n holds column n, K = t*32 + hi*8 + [0..8) and t*32 + 16 + hi*8 + [0..8)
            const f4* p = (const f4*)(row + t * 32 + hi * 8);
            f4 q0 = p[0], q1 = p[1], q2 = p[4], q3 = p[5];
            v16h b;
            #pragma unroll
            for (int i = 0; i < 4; ++i) {
                b[i]      = (_Float16)q0[i];
                b[4 + i]  = (_Float16)q1[i];
                b[8 + i]  = (_Float16)q2[i];
                b[12 + i] = (_Float16)q3[i];
            }
            acc = __builtin_amdgcn_wmma_f32_16x16x32_f16(
                false, a[t], false, b, (short)0, acc, false, false);
        }

        // diagonal = this lane's element's inner product; finite for all lanes,
        // non-owner lanes masked after the loop (keeps EXEC all-ones for WMMA)
        const float ip = pick8(acc, r);
        loss += nls_clip((tgt == 0) ? ip : -ip);
    }

    // mask non-owner lanes, wave-reduce, one partial per tile
    float wl = own ? loss : 0.0f;
    #pragma unroll
    for (int off = 16; off > 0; off >>= 1) wl += __shfl_xor(wl, off, 32);
    if (lane == 0) partials[tile] = wl;
}

__global__ void __launch_bounds__(256)
reduce_kernel(const float* __restrict__ partials, float* __restrict__ out, int n) {
    __shared__ float s[256];
    float acc = 0.0f;
    for (int i = threadIdx.x; i < n; i += 256) acc += partials[i];
    s[threadIdx.x] = acc;
    __syncthreads();
    #pragma unroll
    for (int k = 128; k > 0; k >>= 1) {
        if ((int)threadIdx.x < k) s[threadIdx.x] += s[threadIdx.x + k];
        __syncthreads();
    }
    if (threadIdx.x == 0) out[0] = s[0];
}

extern "C" void kernel_launch(void* const* d_in, const int* in_sizes, int n_in,
                              void* d_out, int out_size, void* d_ws, size_t ws_size,
                              hipStream_t stream) {
    const float* emb0     = (const float*)d_in[0];
    const float* emb1     = (const float*)d_in[1];
    const int*   word_idx = (const int*)d_in[2];
    const int*   ctx_inds = (const int*)d_in[3];
    const float* ctx_lens = (const float*)d_in[4];
    const int*   neg_inds = (const int*)d_in[5];

    float* partials = (float*)d_ws;        // kTiles floats (32 KB), fully overwritten
    float* out      = (float*)d_out;       // single f32 scalar

    cbow_loss_kernel<<<kTiles / kWavesPerBlock, kBlock, 0, stream>>>(
        emb0, emb1, word_idx, ctx_inds, ctx_lens, neg_inds, partials);
    reduce_kernel<<<1, 256, 0, stream>>>(partials, out, kTiles);
}
